// GPT2MoEBlock_35871566856571
// MI455X (gfx1250) — compile-verified, matrix operand order checked by reading, not run on hardware
//
#include <hip/hip_runtime.h>

// ---------------------------------------------------------------------------
// GPT-2 MoE block for gfx1250 (MI455X), wave32 + WMMA bf16 + async-to-LDS.
// B=2, S=1024, D=768, H=12, HD=64, E=8, K=2, F=3072. N_TOK = 2048.
// ---------------------------------------------------------------------------

#define NTOK   2048
#define DMODEL 768
#define NHEAD  12
#define HDIM   64
#define SEQ    1024
#define NEXP   8
#define FDIM   3072

typedef __attribute__((ext_vector_type(16))) __bf16 v16bf;
typedef __attribute__((ext_vector_type(8)))  float  v8f;
typedef __attribute__((ext_vector_type(4)))  int    v4i;

union FragAB {
    v16bf v;
    uint4 q[2];
};

__device__ __forceinline__ unsigned short f32_to_bf16(float f) {
    union { float f; unsigned u; } x; x.f = f;
    unsigned r = x.u + 0x7FFFu + ((x.u >> 16) & 1u);   // RNE
    return (unsigned short)(r >> 16);
}
__device__ __forceinline__ float bf16_to_f32(unsigned short h) {
    union { unsigned u; float f; } x; x.u = ((unsigned)h) << 16;
    return x.f;
}

__device__ __forceinline__ v8f wmma_bf16(const FragAB& a, const FragAB& b, v8f c) {
    return __builtin_amdgcn_wmma_f32_16x16x32_bf16(
        false, a.v, false, b.v, (short)0, c, false, false);
}

// --- async global->LDS (ASYNCcnt path), with sync fallback -----------------
#define HAVE_ASYNC_LDS (__has_builtin(__builtin_amdgcn_global_load_async_to_lds_b128))
#if defined(__AMDGCN__) && !HAVE_ASYNC_LDS
#warning "gfx1250 async-to-LDS builtin unavailable on device pass; using sync fallback"
#endif

__device__ __forceinline__ void async_copy16(const unsigned short* g,
                                             unsigned short* l) {
#if HAVE_ASYNC_LDS
    __builtin_amdgcn_global_load_async_to_lds_b128(
        (__attribute__((address_space(1))) v4i*)g,
        (__attribute__((address_space(3))) v4i*)l, 0, 0);
#else
    *(uint4*)l = *(const uint4*)g;
#endif
}

__device__ __forceinline__ void wait_async0() {
#if __has_builtin(__builtin_amdgcn_s_wait_asynccnt)
    __builtin_amdgcn_s_wait_asynccnt(0);
#elif defined(__AMDGCN__)
    asm volatile("s_wait_asynccnt 0x0" ::: "memory");
#endif
}

// ---------------------------------------------------------------------------
// LayerNorm (f32 in) -> bf16 out.  One block per row of 768.
// ---------------------------------------------------------------------------
__global__ __launch_bounds__(256) void ln_bf16_kernel(
    const float* __restrict__ x, const float* __restrict__ g,
    const float* __restrict__ b, unsigned short* __restrict__ out) {
    const int row = blockIdx.x;
    const int tid = threadIdx.x;
    const float* xr = x + (size_t)row * DMODEL;
    float v0 = xr[tid], v1 = xr[tid + 256], v2 = xr[tid + 512];
    float s  = v0 + v1 + v2;
    float ss = v0 * v0 + v1 * v1 + v2 * v2;
    #pragma unroll
    for (int off = 16; off; off >>= 1) {
        s  += __shfl_xor(s,  off);
        ss += __shfl_xor(ss, off);
    }
    __shared__ float sb[8], ssb[8];
    const int wid = tid >> 5, lane = tid & 31;
    if (lane == 0) { sb[wid] = s; ssb[wid] = ss; }
    __syncthreads();
    if (tid == 0) {
        float ts = 0.f, tss = 0.f;
        #pragma unroll
        for (int i = 0; i < 8; ++i) { ts += sb[i]; tss += ssb[i]; }
        sb[0] = ts; ssb[0] = tss;
    }
    __syncthreads();
    const float mean = sb[0] * (1.0f / DMODEL);
    const float var  = ssb[0] * (1.0f / DMODEL) - mean * mean;
    const float inv  = rsqrtf(var + 1e-5f);
    unsigned short* orow = out + (size_t)row * DMODEL;
    orow[tid]       = f32_to_bf16((v0 - mean) * inv * g[tid]       + b[tid]);
    orow[tid + 256] = f32_to_bf16((v1 - mean) * inv * g[tid + 256] + b[tid + 256]);
    orow[tid + 512] = f32_to_bf16((v2 - mean) * inv * g[tid + 512] + b[tid + 512]);
}

// ---------------------------------------------------------------------------
// Weight conversion + transpose: src f32 [K,N] -> dst bf16 [N,K].
// 32x32 tiles through LDS; block 256 = 32x8.
// ---------------------------------------------------------------------------
__global__ __launch_bounds__(256) void cvt_bf16_t_kernel(
    const float* __restrict__ src, unsigned short* __restrict__ dst,
    int K, int N) {
    __shared__ float tile[32][33];
    const int tx = threadIdx.x & 31, ty = threadIdx.x >> 5;
    const int k0 = blockIdx.y * 32, n0 = blockIdx.x * 32;
    #pragma unroll
    for (int i = 0; i < 32; i += 8)
        tile[ty + i][tx] = src[(size_t)(k0 + ty + i) * N + n0 + tx];
    __syncthreads();
    #pragma unroll
    for (int i = 0; i < 32; i += 8)
        dst[(size_t)(n0 + ty + i) * K + k0 + tx] = f32_to_bf16(tile[tx][ty + i]);
}

// ---------------------------------------------------------------------------
// Router: logits = x2 @ w_router^T ; softmax over 8; top-2 normalized gates.
// ---------------------------------------------------------------------------
__global__ __launch_bounds__(256) void router_kernel(
    const unsigned short* __restrict__ xb, const float* __restrict__ wr,
    float* __restrict__ gates) {
    const int wid = threadIdx.x >> 5, lane = threadIdx.x & 31;
    const int t = blockIdx.x * 8 + wid;
    float acc[NEXP];
    #pragma unroll
    for (int e = 0; e < NEXP; ++e) acc[e] = 0.f;
    const unsigned short* xr = xb + (size_t)t * DMODEL;
    for (int d = lane; d < DMODEL; d += 32) {
        const float xv = bf16_to_f32(xr[d]);
        #pragma unroll
        for (int e = 0; e < NEXP; ++e) acc[e] += xv * wr[e * DMODEL + d];
    }
    #pragma unroll
    for (int off = 16; off; off >>= 1) {
        #pragma unroll
        for (int e = 0; e < NEXP; ++e) acc[e] += __shfl_xor(acc[e], off);
    }
    if (lane == 0) {
        float mx = acc[0];
        #pragma unroll
        for (int e = 1; e < NEXP; ++e) mx = fmaxf(mx, acc[e]);
        float p[NEXP]; float den = 0.f;
        #pragma unroll
        for (int e = 0; e < NEXP; ++e) { p[e] = __expf(acc[e] - mx); den += p[e]; }
        const float rden = 1.0f / den;
        #pragma unroll
        for (int e = 0; e < NEXP; ++e) p[e] *= rden;
        int i0 = 0;
        #pragma unroll
        for (int e = 1; e < NEXP; ++e) if (p[e] > p[i0]) i0 = e;
        int i1 = (i0 == 0) ? 1 : 0;
        #pragma unroll
        for (int e = 0; e < NEXP; ++e) if (e != i0 && p[e] > p[i1]) i1 = e;
        const float s2 = 1.0f / (p[i0] + p[i1]);
        #pragma unroll
        for (int e = 0; e < NEXP; ++e) gates[t * NEXP + e] = 0.f;
        gates[t * NEXP + i0] = p[i0] * s2;
        gates[t * NEXP + i1] = p[i1] * s2;
    }
}

// ---------------------------------------------------------------------------
// Tiled WMMA GEMM with async-to-LDS double buffering.
// C[M,N] = A[M,K](bf16,row-major) * Bt[N,K](bf16, pre-transposed) + bias.
// Block = 256 threads = 8 waves (2x4). Tile 128x128, BK=32.
//   epilogue 1: QKV scatter (Q scaled 1/8; V stored transposed [B,H,HD,S])
//   epilogue 2: + residual -> hs(f32) and d_out(f32)
//   epilogue 3: GELU -> bf16
//   epilogue 4: d_out += gate[t,e] * (acc + bias)
// ---------------------------------------------------------------------------
__global__ __launch_bounds__(256) void gemm_wmma_kernel(
    const unsigned short* __restrict__ A, int lda,
    const unsigned short* __restrict__ Bt, int ldb,
    const float* __restrict__ bias, int ktiles, int epilogue,
    const float* __restrict__ res,
    float* __restrict__ out_f, float* __restrict__ out_f2,
    unsigned short* __restrict__ out_b,
    unsigned short* __restrict__ qb, unsigned short* __restrict__ kb,
    unsigned short* __restrict__ vb,
    const float* __restrict__ gates, int expert, int ncols) {

    const int nb = blockIdx.x * 128;
    const int mb = blockIdx.y * 128;
    __shared__ unsigned short As[2][128][40];
    __shared__ unsigned short Bs[2][128][40];

    const int tid  = threadIdx.x;
    const int lane = tid & 31;
    const int wid  = tid >> 5;
    const int wm   = (wid >> 2) * 64;
    const int wn   = (wid & 3) * 32;
    const int r16  = lane & 15;
    const int half = lane >> 4;
    const int koff = half * 8;

    auto stage = [&](int kt, int buf) {
        #pragma unroll
        for (int i = 0; i < 2; ++i) {
            const int id = tid * 2 + i;
            const int r = id >> 2, c = (id & 3) * 8;
            async_copy16(A  + (size_t)(mb + r) * lda + kt * 32 + c, &As[buf][r][c]);
            async_copy16(Bt + (size_t)(nb + r) * ldb + kt * 32 + c, &Bs[buf][r][c]);
        }
    };

    v8f acc[4][2];
    #pragma unroll
    for (int sm = 0; sm < 4; ++sm)
        #pragma unroll
        for (int sn = 0; sn < 2; ++sn) acc[sm][sn] = (v8f){};

    stage(0, 0);
    wait_async0();
    __syncthreads();

    for (int kt = 0; kt < ktiles; ++kt) {
        const int cur = kt & 1;
        if (kt + 1 < ktiles) stage(kt + 1, cur ^ 1);

        FragAB a[4], b[2];
        #pragma unroll
        for (int sm = 0; sm < 4; ++sm) {
            const int r = wm + sm * 16 + r16;
            a[sm].q[0] = *(const uint4*)&As[cur][r][koff];
            a[sm].q[1] = *(const uint4*)&As[cur][r][16 + koff];
        }
        #pragma unroll
        for (int sn = 0; sn < 2; ++sn) {
            const int n = wn + sn * 16 + r16;
            b[sn].q[0] = *(const uint4*)&Bs[cur][n][koff];
            b[sn].q[1] = *(const uint4*)&Bs[cur][n][16 + koff];
        }
        #pragma unroll
        for (int sm = 0; sm < 4; ++sm)
            #pragma unroll
            for (int sn = 0; sn < 2; ++sn)
                acc[sm][sn] = wmma_bf16(a[sm], b[sn], acc[sm][sn]);

        wait_async0();      // next buffer landed (issued before compute)
        __syncthreads();    // everyone done reading 'cur' before reuse
    }

    // ---- epilogue ----
    #pragma unroll
    for (int sm = 0; sm < 4; ++sm) {
        #pragma unroll
        for (int sn = 0; sn < 2; ++sn) {
            #pragma unroll
            for (int v = 0; v < 8; ++v) {
                const int rg = mb + wm + sm * 16 + v + half * 8;   // token
                const int cg = nb + wn + sn * 16 + r16;            // col
                float val = acc[sm][sn][v] + (bias ? bias[cg] : 0.f);
                if (epilogue == 1) {
                    const int which = cg / DMODEL;
                    const int d  = cg - which * DMODEL;
                    const int h  = d >> 6, hd = d & 63;
                    const int bt = rg >> 10, s = rg & 1023;
                    if (which == 0) {
                        qb[(((size_t)bt * NHEAD + h) * SEQ + s) * HDIM + hd] =
                            f32_to_bf16(val * 0.125f);
                    } else if (which == 1) {
                        kb[(((size_t)bt * NHEAD + h) * SEQ + s) * HDIM + hd] =
                            f32_to_bf16(val);
                    } else {
                        // V transposed: [B,H,HD,S] so attention stages it
                        // with contiguous async row copies.
                        vb[(((size_t)bt * NHEAD + h) * HDIM + hd) * SEQ + s] =
                            f32_to_bf16(val);
                    }
                } else if (epilogue == 2) {
                    val += res[(size_t)rg * ncols + cg];
                    out_f [(size_t)rg * ncols + cg] = val;
                    out_f2[(size_t)rg * ncols + cg] = val;
                } else if (epilogue == 3) {
                    const float gl = 0.5f * val * (1.0f + erff(val * 0.70710678118654752f));
                    out_b[(size_t)rg * ncols + cg] = f32_to_bf16(gl);
                } else if (epilogue == 4) {
                    out_f[(size_t)rg * ncols + cg] +=
                        gates[rg * NEXP + expert] * val;
                } else {
                    out_f[(size_t)rg * ncols + cg] = val;
                }
            }
        }
    }
}

// ---------------------------------------------------------------------------
// Flash-style causal attention with WMMA + async double-buffered K/V tiles.
// Grid (S/64, B*H); block 128 (4 waves). Each wave owns 16 query rows.
// Q pre-scaled by 1/sqrt(HD); V supplied transposed [B,H,HD,S].
// ---------------------------------------------------------------------------
__global__ __launch_bounds__(128) void attn_kernel(
    const unsigned short* __restrict__ Qb, const unsigned short* __restrict__ Kb,
    const unsigned short* __restrict__ Vt, unsigned short* __restrict__ ctx) {

    const int qt = blockIdx.x;                // query tile 0..15
    const int bh = blockIdx.y;                // b*H+h
    const int tid = threadIdx.x, lane = tid & 31, wid = tid >> 5;
    const int r16 = lane & 15, half = lane >> 4, koff = half * 8;

    __shared__ unsigned short Ksh[2][64][72];         // [key][hd]
    __shared__ unsigned short Vsh[2][64][72];         // [hd][key]
    __shared__ unsigned short Psh[4][16][72];         // per-wave P staging

    const size_t baseK = (size_t)bh * SEQ * HDIM;     // K rows: [s][hd]
    const size_t baseV = (size_t)bh * HDIM * SEQ;     // V rows: [hd][s]

    auto stageKV = [&](int kt, int buf) {
        #pragma unroll
        for (int i = 0; i < 4; ++i) {
            const int id = tid * 4 + i;               // 0..511
            const int r = id >> 3, c = (id & 7) * 8;
            async_copy16(Kb + baseK + (size_t)(kt * 64 + r) * HDIM + c,
                         &Ksh[buf][r][c]);
            async_copy16(Vt + baseV + (size_t)r * SEQ + kt * 64 + c,
                         &Vsh[buf][r][c]);
        }
    };

    // Q fragments (A-layout), loaded once.
    FragAB aq[2];
    {
        const int qrow = qt * 64 + wid * 16 + r16;
        const unsigned short* qp = Qb + baseK + (size_t)qrow * HDIM;
        aq[0].q[0] = *(const uint4*)(qp + koff);
        aq[0].q[1] = *(const uint4*)(qp + 16 + koff);
        aq[1].q[0] = *(const uint4*)(qp + 32 + koff);
        aq[1].q[1] = *(const uint4*)(qp + 48 + koff);
    }

    v8f o[4];
    #pragma unroll
    for (int nt = 0; nt < 4; ++nt) o[nt] = (v8f){};
    float mrow[8], lrow[8];
    #pragma unroll
    for (int v = 0; v < 8; ++v) { mrow[v] = -1e30f; lrow[v] = 0.f; }

    stageKV(0, 0);
    wait_async0();
    __syncthreads();

    for (int kt = 0; kt <= qt; ++kt) {
        const int cur = kt & 1;
        if (kt < qt) stageKV(kt + 1, cur ^ 1);

        // scores = Q @ K^T  (16x64 per wave, four 16x16 tiles)
        v8f sc[4];
        #pragma unroll
        for (int nt = 0; nt < 4; ++nt) {
            FragAB bk0, bk1;
            const int n = nt * 16 + r16;
            bk0.q[0] = *(const uint4*)&Ksh[cur][n][koff];
            bk0.q[1] = *(const uint4*)&Ksh[cur][n][16 + koff];
            bk1.q[0] = *(const uint4*)&Ksh[cur][n][32 + koff];
            bk1.q[1] = *(const uint4*)&Ksh[cur][n][48 + koff];
            v8f z = (v8f){};
            z = wmma_bf16(aq[0], bk0, z);
            z = wmma_bf16(aq[1], bk1, z);
            sc[nt] = z;
        }

        if (kt == qt) {   // causal mask on the diagonal tile
            #pragma unroll
            for (int nt = 0; nt < 4; ++nt) {
                const int cg = nt * 16 + r16;
                #pragma unroll
                for (int v = 0; v < 8; ++v) {
                    const int rg = wid * 16 + v + half * 8;
                    if (cg > rg) sc[nt][v] = -1e30f;
                }
            }
        }

        // Online softmax per row.
        #pragma unroll
        for (int v = 0; v < 8; ++v) {
            float mx = fmaxf(fmaxf(sc[0][v], sc[1][v]), fmaxf(sc[2][v], sc[3][v]));
            #pragma unroll
            for (int off = 8; off; off >>= 1) mx = fmaxf(mx, __shfl_xor(mx, off));
            const float mnew  = fmaxf(mrow[v], mx);
            const float scale = __expf(mrow[v] - mnew);
            float rs = 0.f;
            #pragma unroll
            for (int nt = 0; nt < 4; ++nt) {
                const float p = __expf(sc[nt][v] - mnew);
                sc[nt][v] = p;
                rs += p;
            }
            #pragma unroll
            for (int off = 8; off; off >>= 1) rs += __shfl_xor(rs, off);
            lrow[v] = lrow[v] * scale + rs;
            mrow[v] = mnew;
            #pragma unroll
            for (int nt = 0; nt < 4; ++nt) o[nt][v] *= scale;
        }

        // P -> LDS (C-layout) then reload as A-layout bf16 fragments.
        #pragma unroll
        for (int nt = 0; nt < 4; ++nt)
            #pragma unroll
            for (int v = 0; v < 8; ++v)
                Psh[wid][v + half * 8][nt * 16 + r16] = f32_to_bf16(sc[nt][v]);
        asm volatile("s_wait_dscnt 0x0" ::: "memory");   // wave-local LDS RAW

        FragAB ap[2];
        {
            const unsigned short* pr = &Psh[wid][r16][0];
            ap[0].q[0] = *(const uint4*)(pr + koff);
            ap[0].q[1] = *(const uint4*)(pr + 16 + koff);
            ap[1].q[0] = *(const uint4*)(pr + 32 + koff);
            ap[1].q[1] = *(const uint4*)(pr + 48 + koff);
        }

        // O += P @ V   (V tile is [hd][key] so B-fragments are contiguous)
        #pragma unroll
        for (int nt = 0; nt < 4; ++nt) {
            FragAB bv0, bv1;
            const int n = nt * 16 + r16;
            bv0.q[0] = *(const uint4*)&Vsh[cur][n][koff];
            bv0.q[1] = *(const uint4*)&Vsh[cur][n][16 + koff];
            bv1.q[0] = *(const uint4*)&Vsh[cur][n][32 + koff];
            bv1.q[1] = *(const uint4*)&Vsh[cur][n][48 + koff];
            o[nt] = wmma_bf16(ap[0], bv0, o[nt]);
            o[nt] = wmma_bf16(ap[1], bv1, o[nt]);
        }

        wait_async0();
        __syncthreads();
    }

    // Final normalization + scatter to ctx [B,S,D] bf16.
    const int bidx = bh / NHEAD, h = bh % NHEAD;
    #pragma unroll
    for (int nt = 0; nt < 4; ++nt) {
        #pragma unroll
        for (int v = 0; v < 8; ++v) {
            const int rg = qt * 64 + wid * 16 + v + half * 8;
            const int cg = nt * 16 + r16;
            const float val = o[nt][v] / lrow[v];
            ctx[((size_t)(bidx * SEQ + rg)) * DMODEL + h * HDIM + cg] =
                f32_to_bf16(val);
        }
    }
}

// ---------------------------------------------------------------------------
// Host orchestration.
// ---------------------------------------------------------------------------
extern "C" void kernel_launch(void* const* d_in, const int* in_sizes, int n_in,
                              void* d_out, int out_size, void* d_ws, size_t ws_size,
                              hipStream_t stream) {
    (void)in_sizes; (void)n_in; (void)out_size; (void)ws_size;

    const float* hidden     = (const float*)d_in[0];
    const float* ln1_g      = (const float*)d_in[1];
    const float* ln1_b      = (const float*)d_in[2];
    const float* w_attn     = (const float*)d_in[3];
    const float* b_attn     = (const float*)d_in[4];
    const float* w_attnproj = (const float*)d_in[5];
    const float* b_attnproj = (const float*)d_in[6];
    const float* ln2_g      = (const float*)d_in[7];
    const float* ln2_b      = (const float*)d_in[8];
    const float* w_router   = (const float*)d_in[9];
    const float* w_fc       = (const float*)d_in[10];
    const float* b_fc       = (const float*)d_in[11];
    const float* w_proj     = (const float*)d_in[12];
    const float* b_proj     = (const float*)d_in[13];
    float* out = (float*)d_out;

    // Workspace carve-up.
    char* p = (char*)d_ws;
    unsigned short* xb  = (unsigned short*)p; p += (size_t)NTOK * DMODEL * 2;   // LN out
    unsigned short* qb  = (unsigned short*)p; p += (size_t)NTOK * DMODEL * 2;   // Q [B,H,S,HD]
    unsigned short* kb  = (unsigned short*)p; p += (size_t)NTOK * DMODEL * 2;   // K [B,H,S,HD]
    unsigned short* vb  = (unsigned short*)p; p += (size_t)NTOK * DMODEL * 2;   // V [B,H,HD,S]
    unsigned short* wb  = (unsigned short*)p; p += (size_t)DMODEL * FDIM * 2;   // staged weight^T
    unsigned short* ctx = (unsigned short*)p; p += (size_t)NTOK * DMODEL * 2;   // attn ctx bf16
    float*          hs  = (float*)p;          p += (size_t)NTOK * DMODEL * 4;   // post-attn hidden
    unsigned short* he  = (unsigned short*)p; p += (size_t)NTOK * FDIM * 2;     // expert hidden
    float*          gts = (float*)p;          p += (size_t)NTOK * NEXP * 4;     // gates

    // 1) LN1 -> bf16
    ln_bf16_kernel<<<NTOK, 256, 0, stream>>>(hidden, ln1_g, ln1_b, xb);

    // 2) QKV GEMM (B transposed at conversion: [3D, D])
    cvt_bf16_t_kernel<<<dim3(3 * DMODEL / 32, DMODEL / 32), 256, 0, stream>>>(
        w_attn, wb, DMODEL, 3 * DMODEL);
    gemm_wmma_kernel<<<dim3((3 * DMODEL) / 128, NTOK / 128), 256, 0, stream>>>(
        xb, DMODEL, wb, DMODEL, b_attn, DMODEL / 32, 1,
        nullptr, nullptr, nullptr, nullptr, qb, kb, vb, nullptr, 0, 3 * DMODEL);

    // 3) Flash attention
    attn_kernel<<<dim3(SEQ / 64, 2 * NHEAD), 128, 0, stream>>>(qb, kb, vb, ctx);

    // 4) Attention output proj + residual -> hs and d_out
    cvt_bf16_t_kernel<<<dim3(DMODEL / 32, DMODEL / 32), 256, 0, stream>>>(
        w_attnproj, wb, DMODEL, DMODEL);
    gemm_wmma_kernel<<<dim3(DMODEL / 128, NTOK / 128), 256, 0, stream>>>(
        ctx, DMODEL, wb, DMODEL, b_attnproj, DMODEL / 32, 2,
        hidden, hs, out, nullptr, nullptr, nullptr, nullptr, nullptr, 0, DMODEL);

    // 5) LN2 -> bf16
    ln_bf16_kernel<<<NTOK, 256, 0, stream>>>(hs, ln2_g, ln2_b, xb);

    // 6) Router gates
    router_kernel<<<NTOK / 8, 256, 0, stream>>>(xb, w_router, gts);

    // 7) Experts: fc+GELU then proj with gated accumulation into d_out
    for (int e = 0; e < NEXP; ++e) {
        cvt_bf16_t_kernel<<<dim3(FDIM / 32, DMODEL / 32), 256, 0, stream>>>(
            w_fc + (size_t)e * DMODEL * FDIM, wb, DMODEL, FDIM);
        gemm_wmma_kernel<<<dim3(FDIM / 128, NTOK / 128), 256, 0, stream>>>(
            xb, DMODEL, wb, DMODEL, b_fc + (size_t)e * FDIM, DMODEL / 32, 3,
            nullptr, nullptr, nullptr, he, nullptr, nullptr, nullptr, nullptr, 0, FDIM);
        cvt_bf16_t_kernel<<<dim3(DMODEL / 32, FDIM / 32), 256, 0, stream>>>(
            w_proj + (size_t)e * FDIM * DMODEL, wb, FDIM, DMODEL);
        gemm_wmma_kernel<<<dim3(DMODEL / 128, NTOK / 128), 256, 0, stream>>>(
            he, FDIM, wb, FDIM, b_proj + (size_t)e * DMODEL, FDIM / 32, 4,
            nullptr, out, nullptr, nullptr, nullptr, nullptr, nullptr, gts, e, DMODEL);
    }
}
